// MultiHeadScanTISABiasedAttention_59047210385813
// MI455X (gfx1250) — matrix operand order repaired and checked by reading, not compiled, not run on hardware
//
#include <hip/hip_runtime.h>
#include <hip/hip_bf16.h>
#include <math.h>

typedef _Float16 f16;
typedef __attribute__((ext_vector_type(16))) _Float16 v16h;
typedef __attribute__((ext_vector_type(8)))  float    v8f;
typedef __attribute__((ext_vector_type(4)))  int      v4i;

#define B_  2
#define SEQ 2048
#define DM_ 512
#define H_  8
#define F_  5
#define DH_ 64

// ---------------------------------------------------------------------------
// CDNA5 async global->LDS staging (gfx1250 ASYNCcnt path). The builtin exists
// on this toolchain; parameter pointee type is a 4 x i32 vector.
// ---------------------------------------------------------------------------
#if defined(__has_builtin)
#  if __has_builtin(__builtin_amdgcn_global_load_async_to_lds_b128)
#    define USE_ASYNC_LDS 1
#  endif
#endif
#ifndef USE_ASYNC_LDS
#  define USE_ASYNC_LDS 0
#endif

#if USE_ASYNC_LDS
typedef __attribute__((address_space(1))) v4i gv4i;
typedef __attribute__((address_space(3))) v4i lv4i;
#endif

__device__ __forceinline__ void cp16(f16* dst, const f16* src) {
#if USE_ASYNC_LDS
  __builtin_amdgcn_global_load_async_to_lds_b128(
      (gv4i*)(uintptr_t)src, (lv4i*)(unsigned)(uintptr_t)dst, 0, 0);
#else
  *(uint4*)dst = *(const uint4*)src;
#endif
}
__device__ __forceinline__ void cp_wait() {
#if USE_ASYNC_LDS
#  if __has_builtin(__builtin_amdgcn_s_wait_asynccnt)
  __builtin_amdgcn_s_wait_asynccnt(0);
#  else
  asm volatile("s_wait_asynccnt 0" ::: "memory");
#  endif
#endif
}

// ---------------------------------------------------------------------------
// WMMA helpers (CDNA5 gfx1250, wave32). D = A(16x32 f16) * B(32x16 f16) + C(f32)
// ---------------------------------------------------------------------------
__device__ __forceinline__ v8f wmma16(v16h a, v16h b, v8f c) {
  return __builtin_amdgcn_wmma_f32_16x16x32_f16(false, a, false, b, (short)0, c,
                                                false, false);
}

// A fragment from row-major [16 x >=32] tile (f16). ISA layout:
// lane row = lane&15; lanes<16 hold K = 0..7,16..23 ; lanes>=16 hold K = 8..15,24..31
__device__ __forceinline__ v16h a_frag(const f16* base, int ld, int lane) {
  const int row = lane & 15, half = lane >> 4;
  const f16* p = base + row * ld + half * 8;
  v16h a;
#pragma unroll
  for (int i = 0; i < 8; ++i) a[i] = p[i];
#pragma unroll
  for (int i = 0; i < 8; ++i) a[8 + i] = p[16 + i];
  return a;
}

// B fragment where the tile is stored n-major: B[k][n] = base[n*ld + k].
// lane col = lane&15 (n); lanes<16 K=0..7,16..23, lanes>=16 K=8..15,24..31.
__device__ __forceinline__ v16h bT_frag(const f16* base, int ld, int lane) {
  const int col = lane & 15, kb = (lane >> 4) * 8;
  const f16* p = base + col * ld + kb;
  v16h b;
#pragma unroll
  for (int i = 0; i < 8; ++i) b[i] = p[i];
#pragma unroll
  for (int i = 0; i < 8; ++i) b[8 + i] = p[16 + i];
  return b;
}

// ---------------------------------------------------------------------------
// One-time weight transpose: WT_f16[n][k] = W_f32[k][n]   (512x512)
// ---------------------------------------------------------------------------
__global__ __launch_bounds__(256) void transpose_w_kernel(
    const float* __restrict__ W, f16* __restrict__ WT) {
  __shared__ float t[32][33];
  const int tx = threadIdx.x, ty = threadIdx.y;
  const int x = blockIdx.x * 32 + tx;   // source col (n)
  const int y0 = blockIdx.y * 32;       // source row (k) base
#pragma unroll
  for (int i = ty; i < 32; i += 8) t[i][tx] = W[(size_t)(y0 + i) * DM_ + x];
  __syncthreads();
#pragma unroll
  for (int i = ty; i < 32; i += 8)
    WT[(size_t)(blockIdx.x * 32 + i) * DM_ + y0 + tx] = (f16)t[tx][i];
}

// ---------------------------------------------------------------------------
// Projection GEMM: (X_f32[B*seq,DM] @ W + bias)*scale -> f16, per-head layout.
// WT is the pre-transposed f16 weight (n-major). vtrans: store output d-major
// ([B,H,DH,seq], used for V so attention's PV B-fragments read contiguously).
// Block: 64x64 tile, 128 threads = 4 waves, each wave 32x32 (2x2 WMMA).
// ---------------------------------------------------------------------------
__global__ __launch_bounds__(128) void proj_kernel(
    const float* __restrict__ X, const f16* __restrict__ WT,
    const float* __restrict__ bias, float scale, f16* __restrict__ out,
    int vtrans) {
  __shared__ __align__(16) f16 As[64 * 48];   // [m][k] row-major
  __shared__ __align__(16) f16 BsT[64 * 40];  // [n][k] n-major
  const int tid = threadIdx.x, lane = tid & 31, wave = tid >> 5;
  const int m0 = blockIdx.x * 64, n0 = blockIdx.y * 64;
  const int wr = (wave & 1) * 32, wc = (wave >> 1) * 32;
  v8f acc[2][2] = {};
  for (int k0 = 0; k0 < DM_; k0 += 32) {
    __syncthreads();
#pragma unroll
    for (int it = 0; it < 4; ++it) {  // A tile: 64x32 fp32 -> f16
      const int c = tid + 128 * it, row = c >> 3, c4 = (c & 7) * 4;
      const float4 x = *(const float4*)(X + (size_t)(m0 + row) * DM_ + k0 + c4);
      f16* d = &As[row * 48 + c4];
      d[0] = (f16)x.x; d[1] = (f16)x.y; d[2] = (f16)x.z; d[3] = (f16)x.w;
    }
#pragma unroll
    for (int it = 0; it < 2; ++it) {  // B tile: 64n x 32k f16, 16B copies
      const int c = tid + 128 * it, row = c >> 2, c8 = (c & 3) * 8;
      cp16(&BsT[row * 40 + c8], WT + (size_t)(n0 + row) * DM_ + k0 + c8);
    }
    cp_wait();
    __syncthreads();
    v16h a[2], b[2];
#pragma unroll
    for (int mi = 0; mi < 2; ++mi) a[mi] = a_frag(&As[(wr + mi * 16) * 48], 48, lane);
#pragma unroll
    for (int ni = 0; ni < 2; ++ni) b[ni] = bT_frag(&BsT[(wc + ni * 16) * 40], 40, lane);
#pragma unroll
    for (int mi = 0; mi < 2; ++mi)
#pragma unroll
      for (int ni = 0; ni < 2; ++ni) acc[mi][ni] = wmma16(a[mi], b[ni], acc[mi][ni]);
  }
  const int half = lane >> 4, colb = lane & 15;
#pragma unroll
  for (int mi = 0; mi < 2; ++mi)
#pragma unroll
    for (int ni = 0; ni < 2; ++ni)
#pragma unroll
      for (int j = 0; j < 8; ++j) {
        const int mrow = m0 + wr + mi * 16 + j + 8 * half;
        const int ncol = n0 + wc + ni * 16 + colb;
        const float v = (acc[mi][ni][j] + bias[ncol]) * scale;
        const int bb = mrow >> 11, ss = mrow & (SEQ - 1);
        const int hh = ncol >> 6, dd = ncol & (DH_ - 1);
        const size_t idx =
            vtrans ? (((size_t)(bb * H_ + hh) * DH_) + dd) * SEQ + ss
                   : (((size_t)(bb * H_ + hh) * SEQ) + ss) * DH_ + dd;
        out[idx] = (f16)v;
      }
}

// ---------------------------------------------------------------------------
// Fused flash attention with TISA RBF bias + length masking.
// Grid: (Q/128, H, B). Block 256 = 8 waves; wave w owns 16 query rows.
// K tile row-major [key][d]; V tile d-major [d][key] (contiguous B-frag reads).
// ---------------------------------------------------------------------------
__global__ __launch_bounds__(256) void attn_kernel(
    const f16* __restrict__ qws, const f16* __restrict__ kws,
    const f16* __restrict__ vwsT, const float* __restrict__ qlocs,
    const float* __restrict__ klocs, const float* __restrict__ pa,
    const float* __restrict__ pb, const float* __restrict__ pc,
    const int* __restrict__ vlen, f16* __restrict__ ctx) {
  __shared__ __align__(16) f16 kt[64 * 80];     // [key][d]
  __shared__ __align__(16) f16 vtT[64 * 80];    // [d][key]
  __shared__ __align__(16) f16 pt[8][16 * 80];  // per-wave P transpose scratch
  const int tid = threadIdx.x, lane = tid & 31, wave = tid >> 5;
  const int half = lane >> 4, colb = lane & 15;
  const int b = blockIdx.z, h = blockIdx.y;
  const int q0 = blockIdx.x * 128, qw = q0 + wave * 16;
  const f16* qh = qws + (size_t)(b * H_ + h) * SEQ * DH_;
  const f16* kh = kws + (size_t)(b * H_ + h) * SEQ * DH_;
  const f16* vhT = vwsT + (size_t)(b * H_ + h) * SEQ * DH_;  // [DH][SEQ]

  // Q fragments for this wave's 16 rows (contraction over DH=64 -> 2 k-steps)
  v16h aq[2];
  {
    const f16* p = qh + (size_t)(qw + colb) * DH_ + half * 8;
#pragma unroll
    for (int ks = 0; ks < 2; ++ks) {
      v16h t;
#pragma unroll
      for (int i = 0; i < 8; ++i) t[i] = p[ks * 32 + i];
#pragma unroll
      for (int i = 0; i < 8; ++i) t[8 + i] = p[ks * 32 + 16 + i];
      aq[ks] = t;
    }
  }
  float qlx[8], qly[8];
#pragma unroll
  for (int j = 0; j < 8; ++j) {
    const int r = qw + j + 8 * half;
    qlx[j] = qlocs[((size_t)b * SEQ + r) * 2 + 0];
    qly[j] = qlocs[((size_t)b * SEQ + r) * 2 + 1];
  }
  float ra[F_], rb_[F_], rc[F_];
#pragma unroll
  for (int f = 0; f < F_; ++f) {
    ra[f] = pa[h * F_ + f];
    rb_[f] = fabsf(pb[h * F_ + f]);
    rc[f] = pc[h * F_ + f];
  }
  const int vl = vlen[b];
  float m[8], l[8];
  v8f acc[4] = {};
#pragma unroll
  for (int j = 0; j < 8; ++j) { m[j] = -3.0e38f; l[j] = 0.f; }

  for (int k0 = 0; k0 < SEQ; k0 += 64) {
    __syncthreads();
#pragma unroll
    for (int it = 0; it < 2; ++it) {  // stage K (key-major) and V (d-major)
      const int c = tid + 256 * it, row = c >> 3, c8 = (c & 7) * 8;
      cp16(&kt[row * 80 + c8], kh + (size_t)(k0 + row) * DH_ + c8);
      cp16(&vtT[row * 80 + c8], vhT + (size_t)row * SEQ + k0 + c8);
    }
    if (k0 + 64 < SEQ) __builtin_prefetch(kh + (size_t)(k0 + 64) * DH_, 0, 0);
    cp_wait();
    __syncthreads();

    // S = q . k^T for 16 rows x 64 keys (4 col-tiles, 2 k-steps each)
    v8f s[4];
#pragma unroll
    for (int ct = 0; ct < 4; ++ct) {
      v8f sc = {};
      sc = wmma16(aq[0], bT_frag(&kt[ct * 16 * 80 + 0], 80, lane), sc);
      sc = wmma16(aq[1], bT_frag(&kt[ct * 16 * 80 + 32], 80, lane), sc);
      s[ct] = sc;
    }
    // RBF bias + mask fused in C layout; row-wise online softmax
    float nm[8];
#pragma unroll
    for (int j = 0; j < 8; ++j) nm[j] = m[j];
#pragma unroll
    for (int ct = 0; ct < 4; ++ct) {
      const int kg = k0 + ct * 16 + colb;
      const float kx = klocs[((size_t)b * SEQ + kg) * 2 + 0];
      const float ky = klocs[((size_t)b * SEQ + kg) * 2 + 1];
      const bool valid = kg < vl;
#pragma unroll
      for (int j = 0; j < 8; ++j) {
        const float dx = qlx[j] - kx, dy = qly[j] - ky;
        const float d = sqrtf(dx * dx + dy * dy);
        float bias = 0.f;
#pragma unroll
        for (int f = 0; f < F_; ++f) {
          const float t = d - rc[f];
          bias += ra[f] * __expf(-rb_[f] * t * t);
        }
        const float sv = valid ? (s[ct][j] + bias) : -3.0e38f;
        s[ct][j] = sv;
        nm[j] = fmaxf(nm[j], sv);
      }
    }
#pragma unroll
    for (int off = 1; off < 16; off <<= 1)
#pragma unroll
      for (int j = 0; j < 8; ++j) nm[j] = fmaxf(nm[j], __shfl_xor(nm[j], off, 32));
    float esc[8], rs[8];
#pragma unroll
    for (int j = 0; j < 8; ++j) { esc[j] = __expf(m[j] - nm[j]); m[j] = nm[j]; rs[j] = 0.f; }
#pragma unroll
    for (int ct = 0; ct < 4; ++ct)
#pragma unroll
      for (int j = 0; j < 8; ++j) {
        const float p = __expf(s[ct][j] - nm[j]);
        s[ct][j] = p;
        rs[j] += p;
      }
#pragma unroll
    for (int off = 1; off < 16; off <<= 1)
#pragma unroll
      for (int j = 0; j < 8; ++j) rs[j] += __shfl_xor(rs[j], off, 32);
#pragma unroll
    for (int j = 0; j < 8; ++j) l[j] = l[j] * esc[j] + rs[j];
#pragma unroll
    for (int dt = 0; dt < 4; ++dt)
#pragma unroll
      for (int j = 0; j < 8; ++j) acc[dt][j] *= esc[j];

    // C-layout -> A-layout transpose of P via per-wave LDS (wave-local,
    // DS ops are in-order within a wave; explicit split-counter wait).
#pragma unroll
    for (int ct = 0; ct < 4; ++ct)
#pragma unroll
      for (int j = 0; j < 8; ++j)
        pt[wave][(j + 8 * half) * 80 + ct * 16 + colb] = (f16)s[ct][j];
    asm volatile("s_wait_dscnt 0" ::: "memory");
    const v16h ap0 = a_frag(&pt[wave][0], 80, lane);
    const v16h ap1 = a_frag(&pt[wave][32], 80, lane);
#pragma unroll
    for (int dt = 0; dt < 4; ++dt) {
      acc[dt] = wmma16(ap0, bT_frag(&vtT[(dt * 16) * 80 + 0], 80, lane), acc[dt]);
      acc[dt] = wmma16(ap1, bT_frag(&vtT[(dt * 16) * 80 + 32], 80, lane), acc[dt]);
    }
  }
  float inv[8];
#pragma unroll
  for (int j = 0; j < 8; ++j) inv[j] = 1.f / l[j];
#pragma unroll
  for (int dt = 0; dt < 4; ++dt)
#pragma unroll
    for (int j = 0; j < 8; ++j) {
      const int row = qw + j + 8 * half, d = dt * 16 + colb;
      ctx[((size_t)b * SEQ + row) * DM_ + h * DH_ + d] = (f16)(acc[dt][j] * inv[j]);
    }
}

// ---------------------------------------------------------------------------
// Output projection: out_f32[B*Q,DM] = ctx_f16[B*Q,DM] @ Wo + bo  (WoT f16)
// ---------------------------------------------------------------------------
__global__ __launch_bounds__(128) void outproj_kernel(
    const f16* __restrict__ Xh, const f16* __restrict__ WT,
    const float* __restrict__ bias, float* __restrict__ out) {
  __shared__ __align__(16) f16 As[64 * 48];
  __shared__ __align__(16) f16 BsT[64 * 40];
  const int tid = threadIdx.x, lane = tid & 31, wave = tid >> 5;
  const int m0 = blockIdx.x * 64, n0 = blockIdx.y * 64;
  const int wr = (wave & 1) * 32, wc = (wave >> 1) * 32;
  v8f acc[2][2] = {};
  for (int k0 = 0; k0 < DM_; k0 += 32) {
    __syncthreads();
#pragma unroll
    for (int it = 0; it < 2; ++it) {  // A f16: 16B copies
      const int c = tid + 128 * it, row = c >> 2, c8 = (c & 3) * 8;
      cp16(&As[row * 48 + c8], Xh + (size_t)(m0 + row) * DM_ + k0 + c8);
    }
#pragma unroll
    for (int it = 0; it < 2; ++it) {  // B f16 (n-major): 16B copies
      const int c = tid + 128 * it, row = c >> 2, c8 = (c & 3) * 8;
      cp16(&BsT[row * 40 + c8], WT + (size_t)(n0 + row) * DM_ + k0 + c8);
    }
    cp_wait();
    __syncthreads();
    v16h a[2], b[2];
#pragma unroll
    for (int mi = 0; mi < 2; ++mi) a[mi] = a_frag(&As[(wr + mi * 16) * 48], 48, lane);
#pragma unroll
    for (int ni = 0; ni < 2; ++ni) b[ni] = bT_frag(&BsT[(wc + ni * 16) * 40], 40, lane);
#pragma unroll
    for (int mi = 0; mi < 2; ++mi)
#pragma unroll
      for (int ni = 0; ni < 2; ++ni) acc[mi][ni] = wmma16(a[mi], b[ni], acc[mi][ni]);
  }
  const int half = lane >> 4, colb = lane & 15;
#pragma unroll
  for (int mi = 0; mi < 2; ++mi)
#pragma unroll
    for (int ni = 0; ni < 2; ++ni)
#pragma unroll
      for (int j = 0; j < 8; ++j) {
        const int mrow = m0 + wr + mi * 16 + j + 8 * half;
        const int ncol = n0 + wc + ni * 16 + colb;
        out[(size_t)mrow * DM_ + ncol] = acc[mi][ni][j] + bias[ncol];
      }
}

// ---------------------------------------------------------------------------
extern "C" void kernel_launch(void* const* d_in, const int* in_sizes, int n_in,
                              void* d_out, int out_size, void* d_ws, size_t ws_size,
                              hipStream_t stream) {
  const float* qs = (const float*)d_in[0];
  const float* ks = (const float*)d_in[1];
  const float* vs = (const float*)d_in[2];
  const float* qlocs = (const float*)d_in[3];
  const float* klocs = (const float*)d_in[4];
  const float* Wq = (const float*)d_in[5];
  const float* bq = (const float*)d_in[6];
  const float* Wk = (const float*)d_in[7];
  const float* bk = (const float*)d_in[8];
  const float* Wv = (const float*)d_in[9];
  const float* bv = (const float*)d_in[10];
  const float* Wo = (const float*)d_in[11];
  const float* bo = (const float*)d_in[12];
  const float* pa = (const float*)d_in[13];
  const float* pb = (const float*)d_in[14];
  const float* pc = (const float*)d_in[15];
  const int* vlen = (const int*)d_in[16];
  float* out = (float*)d_out;

  char* ws = (char*)d_ws;
  const size_t headBytes = (size_t)B_ * H_ * SEQ * DH_ * 2;  // 4 MB
  const size_t wtBytes = (size_t)DM_ * DM_ * 2;              // 512 KB
  f16* qws = (f16*)(ws + 0 * headBytes);
  f16* kws = (f16*)(ws + 1 * headBytes);
  f16* vwsT = (f16*)(ws + 2 * headBytes);
  f16* ctx = (f16*)(ws + 3 * headBytes);
  f16* WqT = (f16*)(ws + 4 * headBytes + 0 * wtBytes);
  f16* WkT = (f16*)(ws + 4 * headBytes + 1 * wtBytes);
  f16* WvT = (f16*)(ws + 4 * headBytes + 2 * wtBytes);
  f16* WoT = (f16*)(ws + 4 * headBytes + 3 * wtBytes);

  const dim3 tg(DM_ / 32, DM_ / 32), tb(32, 8);
  transpose_w_kernel<<<tg, tb, 0, stream>>>(Wq, WqT);
  transpose_w_kernel<<<tg, tb, 0, stream>>>(Wk, WkT);
  transpose_w_kernel<<<tg, tb, 0, stream>>>(Wv, WvT);
  transpose_w_kernel<<<tg, tb, 0, stream>>>(Wo, WoT);

  const dim3 gg(B_ * SEQ / 64, DM_ / 64);  // (64, 8)
  const dim3 gb(128);
  proj_kernel<<<gg, gb, 0, stream>>>(qs, WqT, bq, 0.125f, qws, 0);  // 1/sqrt(64)
  proj_kernel<<<gg, gb, 0, stream>>>(ks, WkT, bk, 1.0f, kws, 0);
  proj_kernel<<<gg, gb, 0, stream>>>(vs, WvT, bv, 1.0f, vwsT, 1);
  const dim3 ag(SEQ / 128, H_, B_);
  attn_kernel<<<ag, dim3(256), 0, stream>>>(qws, kws, vwsT, qlocs, klocs, pa, pb,
                                            pc, vlen, ctx);
  outproj_kernel<<<gg, gb, 0, stream>>>(ctx, WoT, bo, out);
}
